// DeformableAttention_28140625724104
// MI455X (gfx1250) — matrix operand order repaired and checked by reading, not compiled, not run on hardware
//
#include <hip/hip_runtime.h>
#include <hip/hip_fp16.h>

typedef __attribute__((ext_vector_type(16))) _Float16 v16h;
typedef __attribute__((ext_vector_type(8)))  _Float16 v8h;
typedef __attribute__((ext_vector_type(8)))  float    v8f;
typedef __attribute__((ext_vector_type(4)))  unsigned int v4u;
typedef __attribute__((ext_vector_type(8)))  int      v8i;
typedef __attribute__((ext_vector_type(4)))  int      v4i;

// Problem constants
#define B_    8
#define H_    32
#define W_    32
#define N_    1024
#define DIM_  256
#define DG_   128
#define NG_   2
#define NH_   8
#define DH_   32
#define RPE_W 63   // 2*H-1

// ---------------------------------------------------------------------------
// Tensor Data Mover support (guarded: __has_builtin differs per pass/toolchain)
// ---------------------------------------------------------------------------
#if defined(__has_builtin)
#  if __has_builtin(__builtin_amdgcn_tensor_load_to_lds)
#    define HAVE_TDM 1
#    if __has_include(<hip/amd_detail/amd_gfx1250_TDM.h>)
#      define TDM_ARGS6 1
#    else
#      define TDM_ARGS6 0
#    endif
#  else
#    define HAVE_TDM 0
#  endif
#else
#  define HAVE_TDM 0
#endif

__device__ inline unsigned int lds_byte_off(const void* p)
{
    // flat LDS aperture: low 32 bits of the flat address are the LDS offset
    return (unsigned int)(size_t)p;
}

#if HAVE_TDM
// 2-D f16 tile DMA: `rows` rows of 32 halves (64B), global row stride
// stride_elts (halves). TDM padding inserts 16B after every 64B stored, so the
// tile lands in LDS with the 80B row pitch ([rows][40]) the b128 fragment
// loads want. D# per CDNA5 ISA ch.8: data_size=1(2B), pad_interval=3 (16 DW),
// pad_amount=3 (4 DW).
__device__ inline void tdm_load_tile32h(unsigned int lds_off, const void* gptr,
                                        unsigned int rows, unsigned int stride_elts)
{
    unsigned long long ga = (unsigned long long)(size_t)gptr;
    v4u g0;
    g0[0] = 1u;                                    // count=1, user mode, no gather
    g0[1] = lds_off;                               // LDS byte address
    g0[2] = (unsigned int)ga;                      // global_addr[31:0]
    g0[3] = (unsigned int)(ga >> 32) | (2u << 30); // global_addr[56:32] | type=2
    v8i g1;
    g1[0] = (int)((1u << 16) | (1u << 20) | (3u << 22) | (3u << 25));
    g1[1] = (int)(32u << 16);                      // tensor_dim0[15:0] = 32
    g1[2] = (int)((rows & 0xFFFFu) << 16);         // dim0 hi=0 | tensor_dim1 lo
    g1[3] = (int)(32u << 16);                      // dim1 hi=0 | tile_dim0 = 32
    g1[4] = (int)(rows & 0xFFFFu);                 // tile_dim1 | tile_dim2=0
    g1[5] = (int)stride_elts;                      // tensor_dim0_stride[31:0]
    g1[6] = 0;
    g1[7] = 0;
    v4i z4 = {0, 0, 0, 0};
#if TDM_ARGS6
    v8i z8 = {0, 0, 0, 0, 0, 0, 0, 0};
    __builtin_amdgcn_tensor_load_to_lds(g0, g1, z4, z4, z8, 0);
#else
    __builtin_amdgcn_tensor_load_to_lds(g0, g1, z4, z4, 0);
#endif
}
#endif

// Stage a [ROWS][32] f16 tile (row stride stride_elts halves) into padded LDS
// [ROWS][40]. TDM path: one DMA issue from wave 0. Fallback: dword loads.
template <int THREADS, int ROWS>
__device__ inline void stage_tile32h(_Float16 (*dst)[40], const _Float16* src,
                                     int stride_elts, int tid)
{
#if HAVE_TDM
    if (tid < 32)
        tdm_load_tile32h(lds_byte_off(&dst[0][0]), src, ROWS, stride_elts);
#else
    #pragma unroll
    for (int it = 0; it < (ROWS * 16) / THREADS; ++it) {
        int i = tid + it * THREADS;
        int r = i >> 4, p = i & 15;
        *(unsigned int*)&dst[r][p * 2] =
            *(const unsigned int*)(src + (size_t)r * stride_elts + p * 2);
    }
#endif
}

// s_wait_tensorcnt needs a compile-time immediate (SOPP SIMM16)
template <int ALLOW>
__device__ inline void tile_wait(int tid)
{
#if HAVE_TDM
    if (tid < 32) __builtin_amdgcn_s_wait_tensorcnt(ALLOW);
#else
    (void)tid;
#endif
}

// ---------------------------------------------------------------------------
// WMMA fragment loads as 2x ds_load_b128 (16B-aligned contiguous runs).
//  A 16x32 f16 (row-major tile, pitch 40 halves = 80B):
//    lane: row = lane&15, halves {h*8..h*8+7} and {h*8+16..h*8+23}
//  B 32x16 f16 from K-major tile (N x K, pitch 40):
//    lane: row = col = lane&15, halves {h*8..h*8+7} and {h*8+16..h*8+23}
// ---------------------------------------------------------------------------
__device__ inline v16h load_fragA(const _Float16* rowp, int half)
{
    v8h lo = *(const v8h*)(rowp + half * 8);
    v8h hi = *(const v8h*)(rowp + half * 8 + 16);
    return __builtin_shufflevector(lo, hi, 0, 1, 2, 3, 4, 5, 6, 7,
                                   8, 9, 10, 11, 12, 13, 14, 15);
}
__device__ inline v16h load_fragBT(const _Float16* rowp, int half)
{
    v8h lo = *(const v8h*)(rowp + half * 8);
    v8h hi = *(const v8h*)(rowp + half * 8 + 16);
    return __builtin_shufflevector(lo, hi, 0, 1, 2, 3, 4, 5, 6, 7,
                                   8, 9, 10, 11, 12, 13, 14, 15);
}

// ---------------------------------------------------------------------------
// One-shot prep kernels (tiny): f32->f16 convert, weight transpose+convert.
// ---------------------------------------------------------------------------
__global__ void convert_f16(const float* __restrict__ src,
                            _Float16* __restrict__ dst, int n)
{
    int i = blockIdx.x * 256 + threadIdx.x;
    if (i < n) dst[i] = (_Float16)src[i];
}

__global__ void transpose_convert(const float* __restrict__ W,
                                  _Float16* __restrict__ WT, int K, int Nc)
{
    int i = blockIdx.x * 256 + threadIdx.x;
    if (i < K * Nc) {
        int k = i / Nc, n = i - k * Nc;
        WT[(size_t)n * K + k] = (_Float16)W[i];
    }
}

// ---------------------------------------------------------------------------
// GEMM: C[M,NC] = A[M,KD](f16) * BT[NC,KD](f16), f32 accumulate.
// Block = 256 threads (8 waves), tile 32x64 of C, K-step 32, double-buffered
// TDM tile pipeline.  Epilogues:
//   EPI 0: Cf += bias (f32 out)
//   EPI 1: Cf (f32) and Ch (f16) copies
//   EPI 2: kv split -> Ch = K rows [B,N,DIM] f16; Cv = V transposed [B,DIM,N] f16
// ---------------------------------------------------------------------------
template <int KD, int NC, int EPI>
__global__ __launch_bounds__(256) void gemm_f16wmma(
    const _Float16* __restrict__ A, const _Float16* __restrict__ BT,
    const float* __restrict__ bias, float* __restrict__ Cf,
    _Float16* __restrict__ Ch, _Float16* __restrict__ Cv)
{
    __shared__ __attribute__((aligned(16))) _Float16 Ash[2][32][40];
    __shared__ __attribute__((aligned(16))) _Float16 Bsh[2][64][40];

    const int tid  = threadIdx.x;
    const int wave = tid >> 5;
    const int lane = tid & 31;
    const int half = lane >> 4;
    const int l15  = lane & 15;

    const int rowBase = blockIdx.x * 32;
    const int colBase = blockIdx.y * 64;
    const int rowOff  = (wave >> 2) * 16;   // 0 / 16
    const int colOff  = (wave & 3) * 16;    // 0 / 16 / 32 / 48

    const _Float16* Abase = A  + (size_t)rowBase * KD;
    const _Float16* Bbase = BT + (size_t)colBase * KD;

    constexpr int NT = KD / 32;
    v8f acc = {};

    stage_tile32h<256, 32>(Ash[0], Abase, KD, tid);
    stage_tile32h<256, 64>(Bsh[0], Bbase, KD, tid);

    for (int t = 0; t < NT; ++t) {
        const int bb = t & 1;
        if (t + 1 < NT) {
            stage_tile32h<256, 32>(Ash[bb ^ 1], Abase + (t + 1) * 32, KD, tid);
            stage_tile32h<256, 64>(Bsh[bb ^ 1], Bbase + (t + 1) * 32, KD, tid);
            tile_wait<2>(tid);      // previous tile landed (TDM is in-order)
        } else {
            tile_wait<0>(tid);      // last tile landed
        }
        __syncthreads();

        v16h a = load_fragA(&Ash[bb][rowOff + l15][0], half);
        v16h b = load_fragBT(&Bsh[bb][colOff + l15][0], half);
        acc = __builtin_amdgcn_wmma_f32_16x16x32_f16(false, a, false, b,
                                                     (short)0, acc, false, false);
        __syncthreads();
    }

    const int col = colBase + colOff + l15;
    if constexpr (EPI == 0) {
        const float bv = bias[col];
        #pragma unroll
        for (int r = 0; r < 8; ++r) {
            int row = rowBase + rowOff + r + half * 8;
            Cf[(size_t)row * NC + col] = acc[r] + bv;
        }
    } else if constexpr (EPI == 1) {
        #pragma unroll
        for (int r = 0; r < 8; ++r) {
            int row = rowBase + rowOff + r + half * 8;
            Cf[(size_t)row * NC + col] = acc[r];
            Ch[(size_t)row * NC + col] = (_Float16)acc[r];
        }
    } else {
        #pragma unroll
        for (int r = 0; r < 8; ++r) {
            int row = rowBase + rowOff + r + half * 8;   // = b*N + n
            if (col < DIM_) {
                Ch[(size_t)row * DIM_ + col] = (_Float16)acc[r];      // K rows
            } else {
                int bb2 = row >> 10, n = row & 1023;
                Cv[((size_t)bb2 * DIM_ + (col - DIM_)) * N_ + n] =    // V^T
                    (_Float16)acc[r];
            }
        }
    }
}

// ---------------------------------------------------------------------------
// Depthwise conv 5x5 + LayerNorm + GELU + offset head + tanh -> pos,
// then bilinear grid-sample of grouped q -> x_sampled (f16).
// One block (128 threads = one per channel) per (batch*group, pixel).
// ---------------------------------------------------------------------------
__device__ inline float blockReduce128(float v, float* red, int c)
{
    red[c] = v;
    __syncthreads();
    #pragma unroll
    for (int s = 64; s > 0; s >>= 1) {
        if (c < s) red[c] += red[c + s];
        __syncthreads();
    }
    float r = red[0];
    __syncthreads();
    return r;
}

__global__ __launch_bounds__(128) void conv_offset_sample(
    const float* __restrict__ q, const float* __restrict__ conv_w,
    const float* __restrict__ conv_b, const float* __restrict__ ln_g,
    const float* __restrict__ ln_b, const float* __restrict__ Woff,
    float* __restrict__ pos, _Float16* __restrict__ xs)
{
    __shared__ float red[128];

    const int c   = threadIdx.x;
    const int bg  = blockIdx.x >> 10;     // 0..15
    const int pix = blockIdx.x & 1023;    // 0..1023
    const int b   = bg >> 1;
    const int g   = bg & 1;
    const int py  = pix >> 5;
    const int px  = pix & 31;

    // Depthwise 5x5 conv, zero pad. xg[bg,c,y,x] = q[b, y*W+x, g*DG+c]
    float off = conv_b[c];
    #pragma unroll
    for (int ky = 0; ky < 5; ++ky) {
        int yy = py + ky - 2;
        if (yy < 0 || yy >= H_) continue;
        #pragma unroll
        for (int kx = 0; kx < 5; ++kx) {
            int xx = px + kx - 2;
            if (xx < 0 || xx >= W_) continue;
            off += q[((size_t)b * N_ + yy * W_ + xx) * DIM_ + g * DG_ + c] *
                   conv_w[c * 25 + ky * 5 + kx];
        }
    }

    // LayerNorm over channels
    float mu  = blockReduce128(off, red, c) * (1.0f / DG_);
    float d   = off - mu;
    float var = blockReduce128(d * d, red, c) * (1.0f / DG_);
    float nrm = d * rsqrtf(var + 1e-5f) * ln_g[c] + ln_b[c];
    // exact GELU
    float ge  = 0.5f * nrm * (1.0f + erff(nrm * 0.70710678118654752f));

    // offset = gelu(off) @ Woff[DG,2]
    float o0 = blockReduce128(ge * Woff[c * 2 + 0], red, c);
    float o1 = blockReduce128(ge * Woff[c * 2 + 1], red, c);

    // tanh * (1/H)*FACTOR, add reference grid
    const float fac = 2.0f / 32.0f;
    float gy = ((py + 0.5f) / 32.0f) * 2.0f - 1.0f;
    float gx = ((px + 0.5f) / 32.0f) * 2.0f - 1.0f;
    float posy = tanhf(o0) * fac + gy;
    float posx = tanhf(o1) * fac + gx;

    if (c == 0) {
        pos[((size_t)bg * N_ + pix) * 2 + 0] = posy;
        pos[((size_t)bg * N_ + pix) * 2 + 1] = posx;
    }

    // Bilinear sample of this channel's q plane at (posx, posy), zero pad
    float xf = (posx + 1.0f) * 0.5f * (W_ - 1);
    float yf = (posy + 1.0f) * 0.5f * (H_ - 1);
    float x0 = floorf(xf), y0 = floorf(yf);
    float acc = 0.0f;
    #pragma unroll
    for (int t = 0; t < 4; ++t) {
        float yi = y0 + (t >> 1);
        float xi = x0 + (t & 1);
        float w = (1.0f - fabsf(xf - xi)) * (1.0f - fabsf(yf - yi));
        if (yi >= 0.0f && yi < (float)H_ && xi >= 0.0f && xi < (float)W_)
            acc += q[((size_t)b * N_ + (int)yi * W_ + (int)xi) * DIM_ + g * DG_ + c] * w;
    }
    xs[((size_t)b * N_ + pix) * DIM_ + g * DG_ + c] = (_Float16)acc;
}

// ---------------------------------------------------------------------------
// Fused flash attention with on-the-fly relative-position bias.
// Block = 128 threads (4 waves) handles (b, head, 32 queries).
// Q/K (row-major) and V (pre-transposed [d][n]) arrive f16 via double-buffered
// TDM straight into the WMMA-layout padded LDS tiles. rpe[head] lives in LDS.
// qk scale folded into the bias pass. Online softmax -> no N x N tensor.
// ---------------------------------------------------------------------------
__device__ inline float bilin63(const float* rp, float dy, float dx)
{
    float xf = (dx + 1.0f) * 0.5f * (RPE_W - 1);
    float yf = (dy + 1.0f) * 0.5f * (RPE_W - 1);
    float x0 = floorf(xf), y0 = floorf(yf);
    float acc = 0.0f;
    #pragma unroll
    for (int t = 0; t < 4; ++t) {
        float yi = y0 + (t >> 1);
        float xi = x0 + (t & 1);
        float w = (1.0f - fabsf(xf - xi)) * (1.0f - fabsf(yf - yi));
        if (yi >= 0.0f && yi < (float)RPE_W && xi >= 0.0f && xi < (float)RPE_W)
            acc += rp[(int)yi * RPE_W + (int)xi] * w;
    }
    return acc;
}

__global__ __launch_bounds__(128) void attn_fused(
    const _Float16* __restrict__ qh, const _Float16* __restrict__ kh,
    const _Float16* __restrict__ vT, const float* __restrict__ pos,
    const float* __restrict__ rpe, _Float16* __restrict__ out)
{
    __shared__ float    rpes[RPE_W * RPE_W];
    __shared__ __attribute__((aligned(16))) _Float16 Qs[32][40];      // q x d
    __shared__ __attribute__((aligned(16))) _Float16 Ks[2][32][40];   // key x d
    __shared__ __attribute__((aligned(16))) _Float16 Vs[2][32][40];   // d x key
    __shared__ __attribute__((aligned(16))) _Float16 Psh[32][40];     // q x key
    __shared__ float    Ssh[32][33];
    __shared__ float    posS[32][2];
    __shared__ float    part[4][32];
    __shared__ float    mrow[32], lrow[32], frow[32];

    const int tid  = threadIdx.x;
    const int wave = tid >> 5;
    const int lane = tid & 31;
    const int half = lane >> 4;
    const int l15  = lane & 15;

    const int qt = blockIdx.x;          // query tile (32 of them)
    const int h  = blockIdx.y;          // head
    const int b  = blockIdx.z;          // batch
    const int g  = h >> 2;              // offset group = h / NHG
    const int qb = qt * 32;

    const int rowOff = (wave >> 1) * 16;   // S/O sub-tile row half
    const int colOff = (wave & 1) * 16;    // S/O sub-tile col half

    const int srow  = tid & 31;            // softmax row for this thread
    const int squad = tid >> 5;            // column quarter

    const _Float16* Qg = qh + ((size_t)b * N_ + qb) * DIM_ + h * DH_;
    const _Float16* Kg = kh + (size_t)b * N_ * DIM_ + h * DH_;
    const _Float16* Vg = vT + ((size_t)b * DIM_ + h * DH_) * N_;

    // rpe map for this head -> LDS
    for (int i = tid; i < RPE_W * RPE_W; i += 128)
        rpes[i] = rpe[(size_t)h * RPE_W * RPE_W + i];

    // Prologue: Q tile + first K/V tiles in flight
    stage_tile32h<128, 32>(Qs, Qg, DIM_, tid);
    stage_tile32h<128, 32>(Ks[0], Kg, DIM_, tid);
    stage_tile32h<128, 32>(Vs[0], Vg, N_, tid);
    if (tid < 32) { mrow[tid] = -1e30f; lrow[tid] = 0.0f; }
    tile_wait<2>(tid);          // Q landed (TDM completes in order)
    __syncthreads();

    v16h aq = load_fragA(&Qs[rowOff + l15][0], half);
    const float scale = 0.17677669529663687f;  // DH^-0.5

    v8f o = {};
    for (int t = 0; t < 32; ++t) {
        const int bb = t & 1;
        if (t + 1 < 32) {
            stage_tile32h<128, 32>(Ks[bb ^ 1], Kg + (size_t)(t + 1) * 32 * DIM_, DIM_, tid);
            stage_tile32h<128, 32>(Vs[bb ^ 1], Vg + (t + 1) * 32, N_, tid);
        }
        if (tid < 32) {
            size_t pbase = ((size_t)(b * NG_ + g) * N_ + t * 32 + tid) * 2;
            posS[tid][0] = pos[pbase + 0];
            posS[tid][1] = pos[pbase + 1];
        }
        if (t + 1 < 32) tile_wait<2>(tid);
        else            tile_wait<0>(tid);
        __syncthreads();

        // S = Q * K^T : B[k=d][n=j] = Ks[j][d] -> K-major fragment
        v16h bk = load_fragBT(&Ks[bb][colOff + l15][0], half);
        v8f zc = {};
        v8f s = __builtin_amdgcn_wmma_f32_16x16x32_f16(false, aq, false, bk,
                                                       (short)0, zc, false, false);

        // scale + bias (bilinear from LDS rpe), spill S sub-tile
        #pragma unroll
        for (int r = 0; r < 8; ++r) {
            int iL = rowOff + r + half * 8;
            int jL = colOff + l15;
            int nq = qb + iL;
            float gy = (((nq >> 5) + 0.5f) / 32.0f) * 2.0f - 1.0f;
            float gx = (((nq & 31) + 0.5f) / 32.0f) * 2.0f - 1.0f;
            float dy = (gy - posS[jL][0]) * 0.5f;
            float dx = (gx - posS[jL][1]) * 0.5f;
            Ssh[iL][jL] = s[r] * scale + bilin63(rpes, dy, dx);
        }
        __syncthreads();

        // Online softmax: 128 threads, each reduces 8 columns of one row
        float mx = -1e30f;
        #pragma unroll
        for (int j = 0; j < 8; ++j)
            mx = fmaxf(mx, Ssh[srow][squad * 8 + j]);
        part[squad][srow] = mx;
        __syncthreads();
        if (tid < 32) {
            float m0 = mrow[tid];
            float mn = fmaxf(fmaxf(part[0][tid], part[1][tid]),
                             fmaxf(part[2][tid], part[3][tid]));
            float m2 = fmaxf(m0, mn);
            mrow[tid] = m2;
            frow[tid] = __expf(m0 - m2);
        }
        __syncthreads();
        float mr = mrow[srow];
        float sum = 0.0f;
        v8h pv;
        #pragma unroll
        for (int j = 0; j < 8; ++j) {
            float p = __expf(Ssh[srow][squad * 8 + j] - mr);
            pv[j] = (_Float16)p;
            sum += p;
        }
        *(v8h*)&Psh[srow][squad * 8] = pv;     // one 16B ds_store
        part[squad][srow] = sum;
        __syncthreads();
        if (tid < 32)
            lrow[tid] = lrow[tid] * frow[tid] +
                        part[0][tid] + part[1][tid] + part[2][tid] + part[3][tid];
        __syncthreads();

        // Rescale O, accumulate O += P * V
        #pragma unroll
        for (int r = 0; r < 8; ++r)
            o[r] *= frow[rowOff + r + half * 8];

        v16h ap = load_fragA(&Psh[rowOff + l15][0], half);
        v16h bv = load_fragBT(&Vs[bb][colOff + l15][0], half);
        o = __builtin_amdgcn_wmma_f32_16x16x32_f16(false, ap, false, bv,
                                                   (short)0, o, false, false);
        __syncthreads();
    }

    // Normalize, write f16 for the output projection GEMM
    #pragma unroll
    for (int r = 0; r < 8; ++r) {
        int iL = rowOff + r + half * 8;
        out[((size_t)b * N_ + qb + iL) * DIM_ + h * DH_ + colOff + l15] =
            (_Float16)(o[r] / lrow[iL]);
    }
}

// ---------------------------------------------------------------------------
// Launch
// ---------------------------------------------------------------------------
extern "C" void kernel_launch(void* const* d_in, const int* in_sizes, int n_in,
                              void* d_out, int out_size, void* d_ws, size_t ws_size,
                              hipStream_t stream)
{
    const float* x      = (const float*)d_in[0];
    const float* Wq     = (const float*)d_in[1];
    const float* Wkv    = (const float*)d_in[2];
    const float* conv_w = (const float*)d_in[3];
    const float* conv_b = (const float*)d_in[4];
    const float* ln_g   = (const float*)d_in[5];
    const float* ln_b   = (const float*)d_in[6];
    const float* Woff   = (const float*)d_in[7];
    const float* rpe    = (const float*)d_in[8];
    const float* Wout   = (const float*)d_in[9];
    const float* bout   = (const float*)d_in[10];
    float* out = (float*)d_out;

    const size_t NE = (size_t)B_ * N_ * DIM_;   // 2M activations

    // Workspace carve-up
    char* w = (char*)d_ws;
    float*    qf32 = (float*)w;          w += NE * sizeof(float);
    float*    posb = (float*)w;          w += (size_t)B_ * NG_ * N_ * 2 * sizeof(float);
    _Float16* xf16 = (_Float16*)w;       w += NE * sizeof(_Float16);
    _Float16* qf16 = (_Float16*)w;       w += NE * sizeof(_Float16);
    _Float16* xs16 = (_Float16*)w;       w += NE * sizeof(_Float16);
    _Float16* kf16 = (_Float16*)w;       w += NE * sizeof(_Float16);
    _Float16* vT16 = (_Float16*)w;       w += NE * sizeof(_Float16);
    _Float16* ao16 = (_Float16*)w;       w += NE * sizeof(_Float16);
    _Float16* WqT  = (_Float16*)w;       w += (size_t)DIM_ * DIM_ * sizeof(_Float16);
    _Float16* WkvT = (_Float16*)w;       w += (size_t)DIM_ * 2 * DIM_ * sizeof(_Float16);
    _Float16* WoT  = (_Float16*)w;

    const int M = B_ * N_;  // 8192

    // 0) one-shot converts: x -> f16, weights -> transposed f16 [N][K]
    convert_f16<<<dim3((int)((NE + 255) / 256)), dim3(256), 0, stream>>>(x, xf16, (int)NE);
    transpose_convert<<<dim3((DIM_ * DIM_ + 255) / 256), dim3(256), 0, stream>>>(
        Wq, WqT, DIM_, DIM_);
    transpose_convert<<<dim3((DIM_ * 2 * DIM_ + 255) / 256), dim3(256), 0, stream>>>(
        Wkv, WkvT, DIM_, 2 * DIM_);
    transpose_convert<<<dim3((DIM_ * DIM_ + 255) / 256), dim3(256), 0, stream>>>(
        Wout, WoT, DIM_, DIM_);

    // 1) q = x @ Wq  (f32 for conv kernel + f16 for attention)
    gemm_f16wmma<DIM_, DIM_, 1><<<dim3(M / 32, DIM_ / 64), dim3(256), 0, stream>>>(
        xf16, WqT, nullptr, qf32, qf16, nullptr);

    // 2) conv + LN + GELU + offsets -> pos, grid-sample -> x_sampled (f16)
    conv_offset_sample<<<dim3(B_ * NG_ * N_), dim3(128), 0, stream>>>(
        qf32, conv_w, conv_b, ln_g, ln_b, Woff, posb, xs16);

    // 3) kv = x_sampled @ Wkv  -> K rows f16 + V transposed f16
    gemm_f16wmma<DIM_, 2 * DIM_, 2><<<dim3(M / 32, (2 * DIM_) / 64), dim3(256), 0, stream>>>(
        xs16, WkvT, nullptr, nullptr, kf16, vT16);

    // 4) fused flash attention with on-the-fly rpe bias
    attn_fused<<<dim3(N_ / 32, NH_, B_), dim3(128), 0, stream>>>(
        qf16, kf16, vT16, posb, rpe, ao16);

    // 5) out = attn_out @ Wout + bout
    gemm_f16wmma<DIM_, DIM_, 0><<<dim3(M / 32, DIM_ / 64), dim3(256), 0, stream>>>(
        ao16, WoT, bout, out, nullptr, nullptr);
}
